// CrystalRGCNVAE_79637283602845
// MI455X (gfx1250) — matrix operand order repaired and checked by reading, not compiled, not run on hardware
//
#include <hip/hip_runtime.h>
#include <hip/hip_bf16.h>
#include <math.h>

typedef _Float16 v16h __attribute__((ext_vector_type(16)));
typedef _Float16 v8h  __attribute__((ext_vector_type(8)));
typedef float    v8f  __attribute__((ext_vector_type(8)));

#define DEV __device__ __forceinline__

DEV float gelu_f(float x) { return 0.5f * x * (1.0f + erff(x * 0.7071067811865475f)); }
DEV unsigned fkey(float f) { unsigned u = __float_as_uint(f); return (u >> 31) ? ~u : (u | 0x80000000u); }
DEV float fkeyinv(unsigned k) { return __uint_as_float((k >> 31) ? (k & 0x7fffffffu) : ~k); }

// ---------------------------------------------------------------------------
// Generic WMMA GEMM: out = epilogue(A[M,K](f16) @ Wp[K,Nout](f16,packed) + bias)
// 128 threads = 4 waves; each wave computes NT 16x16 f32 tiles.
// Wp packed so lane's v16h B-fragment is one contiguous 32B load:
//   Wp[(((ntile*Ktiles + kt)*32 + lane)*16 + e] = W[kt*32 + 16*(lane>>4) + e][ntile*16 + (lane&15)]
// EPI: 0 = +bias -> f16 store (per-tile, grid.y allowed)
//      1 = GELU+LN -> f16          (workgroup spans full row: Nout == 64*NT, grid.y==1)
//      2 = GELU+LN + add zp[map[m]] -> f16
//      3 = GELU then dot with gvec -> scalar per row
// ---------------------------------------------------------------------------
template <int NT, int EPI>
__global__ __launch_bounds__(128) void gemm_wmma(
    const _Float16* __restrict__ A, const _Float16* __restrict__ Wp,
    const float* __restrict__ bias, int M, int K, int Nout,
    _Float16* __restrict__ outH,
    const float* __restrict__ lng, const float* __restrict__ lnb,
    const float* __restrict__ addT, const int* __restrict__ map,
    const float* __restrict__ gvec, const float* __restrict__ gb,
    float* __restrict__ outS)
{
    constexpr int NC = 64 * NT;
    __shared__ __align__(16) _Float16 smA[16 * 32];
    __shared__ float smRow[16 * NC];
    __shared__ float smPS[16][8];
    __shared__ float smPQ[16][8];
    __shared__ float smMean[16];
    __shared__ float smRstd[16];

    const int tid = threadIdx.x;
    const int w = tid >> 5, lane = tid & 31;
    const int lanelo = lane & 15, hi = lane >> 4;
    const int m0 = blockIdx.x * 16;
    const int Ktiles = K >> 5;
    const int btile0 = (blockIdx.y * 4 + w) * NT;

    v8f acc[NT] = {};

    for (int kt = 0; kt < Ktiles; ++kt) {
        { // stage 16x32 f16 A tile to LDS (each thread: one 8-byte chunk)
            int row = tid >> 3, seg = tid & 7;
            int m = m0 + row;
            unsigned long long v = 0ull;
            if (m < M)
                v = *(const unsigned long long*)(A + ((size_t)m * K + kt * 32 + seg * 4));
            *(unsigned long long*)(&smA[row * 32 + seg * 4]) = v;
        }
        __syncthreads();
        // A fragment in contiguous k order (k = 16*hi + s, same as Wp packing): lane(lo)=row M, hi selects K-half
        int abase = lanelo * 32 + 16 * hi;
        v8h a_lo = *(const v8h*)(&smA[abase]);
        v8h a_hi = *(const v8h*)(&smA[abase + 8]);
        v16h afrag = __builtin_shufflevector(a_lo, a_hi, 0, 1, 2, 3, 4, 5, 6, 7,
                                             8, 9, 10, 11, 12, 13, 14, 15);
#pragma unroll
        for (int j = 0; j < NT; ++j) {
            const v16h bfrag = *(const v16h*)(Wp + (((size_t)(btile0 + j) * Ktiles + kt) * 32 + lane) * 16);
            acc[j] = __builtin_amdgcn_wmma_f32_16x16x32_f16(false, afrag, false, bfrag,
                                                            (short)0, acc[j], false, false);
        }
        __syncthreads();
    }

    if constexpr (EPI == 0) {
#pragma unroll
        for (int j = 0; j < NT; ++j) {
            int nb = (btile0 + j) * 16 + lanelo;
#pragma unroll
            for (int r = 0; r < 8; ++r) {
                int m = m0 + r + 8 * hi;
                if (m < M) outH[(size_t)m * Nout + nb] = (_Float16)(acc[j][r] + bias[nb]);
            }
        }
    } else {
        // stage GELU(acc+bias) rows into LDS (workgroup owns the full row)
#pragma unroll
        for (int j = 0; j < NT; ++j) {
            int nc = (w * NT + j) * 16 + lanelo;
#pragma unroll
            for (int r = 0; r < 8; ++r)
                smRow[(r + 8 * hi) * NC + nc] = gelu_f(acc[j][r] + bias[nc]);
        }
        __syncthreads();
        int rr = tid >> 3, jj = tid & 7;
        if constexpr (EPI == 3) {
            float p = 0.f;
            for (int c = jj; c < NC; c += 8) p += smRow[rr * NC + c] * gvec[c];
            smPS[rr][jj] = p;
            __syncthreads();
            if (tid < 16) {
                float s = 0.f;
                for (int t = 0; t < 8; ++t) s += smPS[tid][t];
                int m = m0 + tid;
                if (m < M) outS[m] = s + gb[0];
            }
        } else {
            float p = 0.f, q = 0.f;
            for (int c = jj; c < NC; c += 8) {
                float v = smRow[rr * NC + c];
                p += v; q += v * v;
            }
            smPS[rr][jj] = p; smPQ[rr][jj] = q;
            __syncthreads();
            if (tid < 16) {
                float s = 0.f, sq = 0.f;
                for (int t = 0; t < 8; ++t) { s += smPS[tid][t]; sq += smPQ[tid][t]; }
                float mean = s / NC;
                float var = sq / NC - mean * mean;
                smMean[tid] = mean;
                smRstd[tid] = rsqrtf(fmaxf(var, 0.f) + 1e-5f);
            }
            __syncthreads();
            for (int idx = tid; idx < 16 * NC; idx += 128) {
                int r = idx / NC, c = idx % NC;
                int m = m0 + r;
                if (m < M) {
                    float v = (smRow[idx] - smMean[r]) * smRstd[r] * lng[c] + lnb[c];
                    if constexpr (EPI == 2) v += addT[(size_t)map[m] * NC + c];
                    outH[(size_t)m * NC + c] = (_Float16)v;
                }
            }
        }
    }
}

// ---------------------------------------------------------------------------
// Prep / elementwise / scatter kernels
// ---------------------------------------------------------------------------
__global__ void k_addvec(float* d, const float* a, const float* b, int n) {
    int i = blockIdx.x * 256 + threadIdx.x;
    if (i < n) d[i] = a[i] + b[i];
}
__global__ void k_fill_f32(float* d, float v, int n) {
    int i = blockIdx.x * 256 + threadIdx.x;
    if (i < n) d[i] = v;
}
__global__ void k_fill_u32(unsigned* d, unsigned v, int n) {
    int i = blockIdx.x * 256 + threadIdx.x;
    if (i < n) d[i] = v;
}

// W_r = sum_b coeff[r,b] * basis[b]  (rgcn1: [4 relations][4][128])
__global__ void k_build_w1r(float* w1r, const float* basis, const float* coeff) {
    int idx = blockIdx.x * 256 + threadIdx.x;
    if (idx >= 4 * 4 * 128) return;
    int r = idx >> 9, rest = idx & 511, i = rest >> 7, c = rest & 127;
    float s = 0.f;
    for (int b = 0; b < 4; ++b) s += coeff[r * 4 + b] * basis[((b * 4 + i) << 7) + c];
    w1r[idx] = s;
}

// rgcn2 combined weight [128 x 1280]: cols [0,1024)=relation tables, [1024,1280)=loop+res
__global__ void k_build_w2(float* W2f, const float* basis, const float* coeff,
                           const float* loopw, const float* resw) {
    int idx = blockIdx.x * 256 + threadIdx.x;
    if (idx >= 128 * 1280) return;
    int k = idx / 1280, n = idx - k * 1280;
    float s;
    if (n < 1024) {
        int r = n >> 8, c = n & 255;
        s = 0.f;
        for (int b = 0; b < 4; ++b)
            s += coeff[r * 4 + b] * basis[((size_t)(b * 128 + k)) * 256 + c];
    } else {
        int c = n - 1024;
        s = loopw[k * 256 + c] + resw[k * 256 + c];
    }
    W2f[idx] = s;
}
__global__ void k_build_b1280(float* b1280, const float* loopb, const float* resb) {
    int n = blockIdx.x * 256 + threadIdx.x;
    if (n >= 1280) return;
    b1280[n] = (n < 1024) ? 0.f : (loopb[n - 1024] + resb[n - 1024]);
}

// Pack row-major f32 [K,Nout] weight -> WMMA-B fragment order f16
__global__ void k_pack_f32(_Float16* dst, const float* src, int K, int Nout) {
    int idx = blockIdx.x * 256 + threadIdx.x;
    if (idx >= K * Nout) return;
    int e = idx & 15, lane = (idx >> 4) & 31, tk = idx >> 9;
    int Ktiles = K >> 5;
    int kt = tk % Ktiles, nt = tk / Ktiles;
    int k = kt * 32 + 16 * (lane >> 4) + e;
    int n = nt * 16 + (lane & 15);
    dst[idx] = (_Float16)src[(size_t)k * Nout + n];
}

// rgcn1: x[N,4] -> per-relation tables (f16) + self part (agg init, f32)
__global__ void k_rgcn1_tables(const float* __restrict__ x, const float* __restrict__ w1r,
                               const float* __restrict__ selfW, const float* __restrict__ selfB,
                               _Float16* __restrict__ tab, float* __restrict__ agg, int N) {
    int idx = blockIdx.x * 256 + threadIdx.x;
    if (idx >= N * 128) return;
    int n = idx >> 7, c = idx & 127;
    float x0 = x[n * 4 + 0], x1 = x[n * 4 + 1], x2 = x[n * 4 + 2], x3 = x[n * 4 + 3];
#pragma unroll
    for (int r = 0; r < 4; ++r) {
        const float* wr = w1r + (r << 9);
        float s = x0 * wr[c] + x1 * wr[128 + c] + x2 * wr[256 + c] + x3 * wr[384 + c];
        tab[(((size_t)n * 4 + r) << 7) + c] = (_Float16)s;
    }
    float sf = x0 * selfW[c] + x1 * selfW[128 + c] + x2 * selfW[256 + c] + x3 * selfW[384 + c];
    agg[((size_t)n << 7) + c] = sf + selfB[c];
}

__global__ void k_scatter1(const _Float16* __restrict__ tab, const int* __restrict__ src,
                           const int* __restrict__ dst, const int* __restrict__ ety,
                           float* __restrict__ agg, int E) {
    long long idx = (long long)blockIdx.x * 256 + threadIdx.x;
    if (idx >= (long long)E * 128) return;
    int e = (int)(idx >> 7), c = (int)(idx & 127);
    float v = (float)tab[(((size_t)src[e] * 4 + ety[e]) << 7) + c];
    atomicAdd(&agg[((size_t)dst[e] << 7) + c], v);
}
__global__ void k_init_agg2(const _Float16* __restrict__ T2, float* __restrict__ agg, int N) {
    long long idx = (long long)blockIdx.x * 256 + threadIdx.x;
    if (idx >= (long long)N * 256) return;
    int n = (int)(idx >> 8), c = (int)(idx & 255);
    agg[idx] = (float)T2[(size_t)n * 1280 + 1024 + c];
}
__global__ void k_scatter2(const _Float16* __restrict__ T2, const int* __restrict__ src,
                           const int* __restrict__ dst, const int* __restrict__ ety,
                           float* __restrict__ agg, int E) {
    long long idx = (long long)blockIdx.x * 256 + threadIdx.x;
    if (idx >= (long long)E * 256) return;
    int e = (int)(idx >> 8), c = (int)(idx & 255);
    float v = (float)T2[(size_t)src[e] * 1280 + (size_t)ety[e] * 256 + c];
    atomicAdd(&agg[((size_t)dst[e] << 8) + c], v);
}

// LN(gelu(a)) rowwise; one wave(32) per row, C in {128,256}
__global__ void k_ln_act(const float* __restrict__ a, int M, int C,
                         const float* __restrict__ g, const float* __restrict__ bt,
                         _Float16* __restrict__ outH, float* __restrict__ outF) {
    int row = blockIdx.x * 4 + (threadIdx.x >> 5);
    int lane = threadIdx.x & 31;
    if (row >= M) return;
    int cnt = C >> 5;
    float yv[8];
    float s = 0.f, q = 0.f;
    for (int i = 0; i < cnt; ++i) {
        int c = lane + (i << 5);
        float y = gelu_f(a[(size_t)row * C + c]);
        yv[i] = y; s += y; q += y * y;
    }
    for (int m = 16; m >= 1; m >>= 1) { s += __shfl_xor(s, m, 32); q += __shfl_xor(q, m, 32); }
    float mean = s / C;
    float rstd = rsqrtf(fmaxf(q / C - mean * mean, 0.f) + 1e-5f);
    for (int i = 0; i < cnt; ++i) {
        int c = lane + (i << 5);
        float v = (yv[i] - mean) * rstd * g[c] + bt[c];
        outH[(size_t)row * C + c] = (_Float16)v;
        if (outF) outF[(size_t)row * C + c] = v;
    }
}

// attention pooling
__global__ void k_gmax(const float* gate, const int* n2g, unsigned* gkey, int N) {
    int n = blockIdx.x * 256 + threadIdx.x;
    if (n < N) atomicMax(&gkey[n2g[n]], fkey(gate[n]));
}
__global__ void k_expsum(const float* gate, const int* n2g, const unsigned* gkey,
                         float* eArr, float* denom, int N) {
    int n = blockIdx.x * 256 + threadIdx.x;
    if (n >= N) return;
    float ev = expf(gate[n] - fkeyinv(gkey[n2g[n]]));
    eArr[n] = ev;
    atomicAdd(&denom[n2g[n]], ev);
}
__global__ void k_pool(const float* __restrict__ h2f, const float* __restrict__ eArr,
                       const float* __restrict__ denom, const int* __restrict__ n2g,
                       float* __restrict__ gemb, int N) {
    long long idx = (long long)blockIdx.x * 256 + threadIdx.x;
    if (idx >= (long long)N * 256) return;
    int n = (int)(idx >> 8), c = (int)(idx & 255);
    int g = n2g[n];
    atomicAdd(&gemb[(size_t)g * 256 + c], (eArr[n] / denom[g]) * h2f[idx]);
}

__global__ void k_mu_logvar_z(const float* __restrict__ gemb,
                              const float* __restrict__ muw, const float* __restrict__ mub,
                              const float* __restrict__ lvw, const float* __restrict__ lvb,
                              const float* __restrict__ eps,
                              float* oMu, float* oLv, float* oZ, float* zbuf, int B) {
    int idx = blockIdx.x * 256 + threadIdx.x;
    if (idx >= B * 64) return;
    int b = idx >> 6, l = idx & 63;
    float am = 0.f, av = 0.f;
    for (int k = 0; k < 256; ++k) {
        float ge = gemb[(size_t)b * 256 + k];
        am += ge * muw[k * 64 + l];
        av += ge * lvw[k * 64 + l];
    }
    float mu = fminf(fmaxf(am + mub[l], -5.f), 5.f);
    float lv = fminf(fmaxf(av + lvb[l], -10.f), 10.f);
    float std = fminf(fmaxf(expf(0.5f * lv), 1e-6f), 1e6f);
    float z = mu + eps[idx] * std;
    oMu[idx] = mu; oLv[idx] = lv; oZ[idx] = z; zbuf[idx] = z;
}

// per-graph MLP head: y = LN(gelu(z @ w1 + b1)); optional out2 = y @ w2 + b2
__global__ void k_latent_head(const float* __restrict__ z, const float* __restrict__ w1,
                              const float* __restrict__ b1, const float* __restrict__ g,
                              const float* __restrict__ bt, int H,
                              const float* __restrict__ w2, const float* __restrict__ b2,
                              int OUTC, float* out1, float* out2) {
    __shared__ float sy[256];
    __shared__ float red[256];
    int b = blockIdx.x, c = threadIdx.x;
    float acc = 0.f;
    for (int k = 0; k < 64; ++k) acc += z[b * 64 + k] * w1[k * H + c];
    float y = gelu_f(acc + b1[c]);
    red[c] = y; __syncthreads();
    for (int s = H >> 1; s > 0; s >>= 1) { if (c < s) red[c] += red[c + s]; __syncthreads(); }
    float mean = red[0] / H; __syncthreads();
    red[c] = (y - mean) * (y - mean); __syncthreads();
    for (int s = H >> 1; s > 0; s >>= 1) { if (c < s) red[c] += red[c + s]; __syncthreads(); }
    float rstd = rsqrtf(red[0] / H + 1e-5f); __syncthreads();
    float yn = (y - mean) * rstd * g[c] + bt[c];
    sy[c] = yn;
    if (out1) out1[(size_t)b * H + c] = yn;
    __syncthreads();
    if (w2 && c < OUTC) {
        float a2 = 0.f;
        for (int h = 0; h < H; ++h) a2 += sy[h] * w2[h * OUTC + c];
        out2[b * OUTC + c] = a2 + b2[c];
    }
}

// pair table: pairZ[p] = concat(zp[p/B], zp[p%B]) as f16
__global__ void k_pairz(const float* __restrict__ zp, _Float16* __restrict__ pz, int B, int P) {
    int idx = blockIdx.x * 256 + threadIdx.x;
    if (idx >= P * 512) return;
    int p = idx >> 9, c = idx & 511;
    int gs = p / B, gd = p - gs * B;
    float v = (c < 256) ? zp[(size_t)gs * 256 + c] : zp[(size_t)gd * 256 + (c - 256)];
    pz[idx] = (_Float16)v;
}

// small output layer: out[M,OUTC] = Ah[M,K](f16) @ w + b
__global__ void k_out_small(const _Float16* __restrict__ Ah, int K, const float* __restrict__ w,
                            const float* __restrict__ b, int M, int OUTC, float* __restrict__ out) {
    long long idx = (long long)blockIdx.x * 256 + threadIdx.x;
    if (idx >= (long long)M * OUTC) return;
    int m = (int)(idx / OUTC), j = (int)(idx - (long long)m * OUTC);
    float acc = 0.f;
    for (int k = 0; k < K; ++k) acc += (float)Ah[(size_t)m * K + k] * w[k * OUTC + j];
    out[idx] = acc + b[j];
}

__global__ void k_edge_write(const int* __restrict__ src, const int* __restrict__ dst,
                             const int* __restrict__ n2g, const float* __restrict__ pairOut,
                             float* __restrict__ oEdge, int E, int B) {
    int e = blockIdx.x * 256 + threadIdx.x;
    if (e >= E) return;
    int p = n2g[src[e]] * B + n2g[dst[e]];
    oEdge[(size_t)e * 3 + 0] = pairOut[(size_t)p * 3 + 0];
    oEdge[(size_t)e * 3 + 1] = pairOut[(size_t)p * 3 + 1];
    oEdge[(size_t)e * 3 + 2] = pairOut[(size_t)p * 3 + 2];
}

// ---------------------------------------------------------------------------
static inline int ceil_div(long long a, int b) { return (int)((a + b - 1) / b); }

extern "C" void kernel_launch(void* const* d_in, const int* in_sizes, int n_in,
                              void* d_out, int out_size, void* d_ws, size_t ws_size,
                              hipStream_t stream) {
    const int N = in_sizes[0] / 4;
    const int E = in_sizes[1];
    const int B = in_sizes[5] / 64;
    const int P = B * B;

    const float* node_feats = (const float*)d_in[0];
    const int* src = (const int*)d_in[1];
    const int* dst = (const int*)d_in[2];
    const int* ety = (const int*)d_in[3];
    const int* n2g = (const int*)d_in[4];
    const float* eps = (const float*)d_in[5];
    auto F = [&](int i) { return (const float*)d_in[i]; };
    // params (flattened in setup_inputs dict insertion order)
    const float *r1_basis = F(6), *r1_coeff = F(7), *r1_lw = F(8), *r1_lb = F(9),
                *r1_rw = F(10), *r1_rb = F(11), *r1_g = F(12), *r1_b = F(13);
    const float *r2_basis = F(14), *r2_coeff = F(15), *r2_lw = F(16), *r2_lb = F(17),
                *r2_rw = F(18), *r2_rb = F(19), *r2_g = F(20), *r2_b = F(21);
    const float *g1w = F(22), *g1b = F(23), *g2w = F(24), *g2b = F(25);
    const float *muw = F(26), *mub = F(27), *lvw = F(28), *lvb = F(29);
    const float *lpw = F(30), *lpb = F(31), *llg = F(32), *llb = F(33);
    const float *nepw = F(34), *nepb = F(35), *nlg = F(36), *nlb = F(37);
    const float *nd1w = F(38), *nd1b = F(39), *ndg = F(40), *ndb = F(41);
    const float *nd2w = F(42), *nd2b = F(43);
    const float *ed1w = F(44), *ed1b = F(45), *e1g = F(46), *e1b = F(47);
    const float *ed2w = F(48), *ed2b = F(49), *e2g = F(50), *e2b = F(51);
    const float *ed3w = F(52), *ed3b = F(53);
    const float *en1w = F(54), *en1b = F(55), *eng = F(56), *enb = F(57);
    const float *en2w = F(58), *en2b = F(59);
    const float *st1w = F(60), *st1b = F(61), *stg = F(62), *stb = F(63);
    const float *st2w = F(64), *st2b = F(65);

    // outputs (f32, tuple order)
    float* out = (float*)d_out;
    float* oMu = out;
    float* oLv = oMu + (size_t)B * 64;
    float* oZ = oLv + (size_t)B * 64;
    float* oNode = oZ + (size_t)B * 64;
    float* oEdge = oNode + (size_t)N * 4;
    float* oEn = oEdge + (size_t)E * 3;
    float* oSt = oEn + (size_t)B * 2;

    // workspace bump allocator (256B aligned)
    size_t off = 0;
    auto alloc = [&](size_t bytes) {
        void* p = (char*)d_ws + off;
        off = (off + bytes + 255) & ~(size_t)255;
        return p;
    };
    float* w1r = (float*)alloc(2048 * 4);
    float* selfW1 = (float*)alloc(512 * 4);
    float* selfB1 = (float*)alloc(128 * 4);
    float* W2f = (float*)alloc((size_t)128 * 1280 * 4);
    _Float16* W2p = (_Float16*)alloc((size_t)128 * 1280 * 2);
    float* b1280 = (float*)alloc(1280 * 4);
    _Float16* g1p = (_Float16*)alloc((size_t)256 * 128 * 2);
    _Float16* nepp = (_Float16*)alloc((size_t)256 * 256 * 2);
    _Float16* nd1p = (_Float16*)alloc((size_t)256 * 128 * 2);
    _Float16* ed1p = (_Float16*)alloc((size_t)512 * 256 * 2);
    _Float16* ed2p = (_Float16*)alloc((size_t)256 * 128 * 2);
    _Float16* tab1 = (_Float16*)alloc((size_t)N * 512 * 2);
    float* agg1 = (float*)alloc((size_t)N * 128 * 4);
    _Float16* h1h = (_Float16*)alloc((size_t)N * 128 * 2);
    _Float16* T2h = (_Float16*)alloc((size_t)N * 1280 * 2);
    float* agg2 = (float*)alloc((size_t)N * 256 * 4);
    _Float16* h2h = (_Float16*)alloc((size_t)N * 256 * 2);
    float* h2f = (float*)alloc((size_t)N * 256 * 4);
    float* gate = (float*)alloc((size_t)N * 4);
    float* eArr = (float*)alloc((size_t)N * 4);
    unsigned* gkey = (unsigned*)alloc((size_t)B * 4);
    float* denom = (float*)alloc((size_t)B * 4);
    float* gemb = (float*)alloc((size_t)B * 256 * 4);
    float* zbuf = (float*)alloc((size_t)B * 64 * 4);
    float* zp = (float*)alloc((size_t)B * 256 * 4);
    _Float16* pairZ = (_Float16*)alloc((size_t)P * 512 * 2);
    _Float16* he1 = (_Float16*)alloc((size_t)P * 256 * 2);
    _Float16* he2 = (_Float16*)alloc((size_t)P * 128 * 2);
    float* pairOut = (float*)alloc((size_t)P * 3 * 4);
    _Float16* nodein = (_Float16*)alloc((size_t)N * 256 * 2);
    _Float16* hn = (_Float16*)alloc((size_t)N * 128 * 2);
    (void)ws_size; (void)n_in; (void)out_size;

    // ---- prep: combined weights, f16 fragment packing ----
    k_build_w1r<<<ceil_div(2048, 256), 256, 0, stream>>>(w1r, r1_basis, r1_coeff);
    k_addvec<<<ceil_div(512, 256), 256, 0, stream>>>(selfW1, r1_lw, r1_rw, 512);
    k_addvec<<<1, 256, 0, stream>>>(selfB1, r1_lb, r1_rb, 128);
    k_build_w2<<<ceil_div((long long)128 * 1280, 256), 256, 0, stream>>>(W2f, r2_basis, r2_coeff, r2_lw, r2_rw);
    k_build_b1280<<<ceil_div(1280, 256), 256, 0, stream>>>(b1280, r2_lb, r2_rb);
    k_pack_f32<<<ceil_div((long long)128 * 1280, 256), 256, 0, stream>>>(W2p, W2f, 128, 1280);
    k_pack_f32<<<ceil_div((long long)256 * 128, 256), 256, 0, stream>>>(g1p, g1w, 256, 128);
    k_pack_f32<<<ceil_div((long long)256 * 256, 256), 256, 0, stream>>>(nepp, nepw, 256, 256);
    k_pack_f32<<<ceil_div((long long)256 * 128, 256), 256, 0, stream>>>(nd1p, nd1w, 256, 128);
    k_pack_f32<<<ceil_div((long long)512 * 256, 256), 256, 0, stream>>>(ed1p, ed1w, 512, 256);
    k_pack_f32<<<ceil_div((long long)256 * 128, 256), 256, 0, stream>>>(ed2p, ed2w, 256, 128);

    // ---- rgcn1: tables + self, edge scatter, LN(gelu) ----
    k_rgcn1_tables<<<ceil_div((long long)N * 128, 256), 256, 0, stream>>>(node_feats, w1r, selfW1, selfB1, tab1, agg1, N);
    k_scatter1<<<ceil_div((long long)E * 128, 256), 256, 0, stream>>>(tab1, src, dst, ety, agg1, E);
    k_ln_act<<<ceil_div(N, 4), 128, 0, stream>>>(agg1, N, 128, r1_g, r1_b, h1h, nullptr);

    // ---- rgcn2: one WMMA GEMM builds all 4 relation tables + self part ----
    gemm_wmma<4, 0><<<dim3(ceil_div(N, 16), 5), 128, 0, stream>>>(
        h1h, W2p, b1280, N, 128, 1280, T2h, nullptr, nullptr, nullptr, nullptr, nullptr, nullptr, nullptr);
    k_init_agg2<<<ceil_div((long long)N * 256, 256), 256, 0, stream>>>(T2h, agg2, N);
    k_scatter2<<<ceil_div((long long)E * 256, 256), 256, 0, stream>>>(T2h, src, dst, ety, agg2, E);
    k_ln_act<<<ceil_div(N, 4), 128, 0, stream>>>(agg2, N, 256, r2_g, r2_b, h2h, h2f);

    // ---- gate head (GEMM + fused gelu+dot), softmax attention pooling ----
    gemm_wmma<2, 3><<<dim3(ceil_div(N, 16), 1), 128, 0, stream>>>(
        h2h, g1p, g1b, N, 256, 128, nullptr, nullptr, nullptr, nullptr, nullptr, g2w, g2b, gate);
    k_fill_u32<<<1, 256, 0, stream>>>(gkey, 0u, B);
    k_fill_f32<<<1, 256, 0, stream>>>(denom, 0.f, B);
    k_fill_f32<<<ceil_div((long long)B * 256, 256), 256, 0, stream>>>(gemb, 0.f, B * 256);
    k_gmax<<<ceil_div(N, 256), 256, 0, stream>>>(gate, n2g, gkey, N);
    k_expsum<<<ceil_div(N, 256), 256, 0, stream>>>(gate, n2g, gkey, eArr, denom, N);
    k_pool<<<ceil_div((long long)N * 256, 256), 256, 0, stream>>>(h2f, eArr, denom, n2g, gemb, N);

    // ---- latent: mu/logvar/z, zp = LN(gelu(latent_proj(z))), energy/stress heads ----
    k_mu_logvar_z<<<ceil_div((long long)B * 64, 256), 256, 0, stream>>>(
        gemb, muw, mub, lvw, lvb, eps, oMu, oLv, oZ, zbuf, B);
    k_latent_head<<<B, 256, 0, stream>>>(zbuf, lpw, lpb, llg, llb, 256, nullptr, nullptr, 0, zp, nullptr);
    k_latent_head<<<B, 128, 0, stream>>>(zbuf, en1w, en1b, eng, enb, 128, en2w, en2b, 2, nullptr, oEn);
    k_latent_head<<<B, 128, 0, stream>>>(zbuf, st1w, st1b, stg, stb, 128, st2w, st2b, 9, nullptr, oSt);

    // ---- edge decoder on the B*B distinct (graph_src, graph_dst) pairs (exact) ----
    k_pairz<<<ceil_div((long long)P * 512, 256), 256, 0, stream>>>(zp, pairZ, B, P);
    gemm_wmma<4, 1><<<dim3(ceil_div(P, 16), 1), 128, 0, stream>>>(
        pairZ, ed1p, ed1b, P, 512, 256, he1, e1g, e1b, nullptr, nullptr, nullptr, nullptr, nullptr);
    gemm_wmma<2, 1><<<dim3(ceil_div(P, 16), 1), 128, 0, stream>>>(
        he1, ed2p, ed2b, P, 256, 128, he2, e2g, e2b, nullptr, nullptr, nullptr, nullptr, nullptr);
    k_out_small<<<ceil_div((long long)P * 3, 256), 256, 0, stream>>>(he2, 128, ed3w, ed3b, P, 3, pairOut);
    k_edge_write<<<ceil_div(E, 256), 256, 0, stream>>>(src, dst, n2g, pairOut, oEdge, E, B);

    // ---- node decoder: nep (GEMM + gelu+LN + z_exp add), nd1, nd2 ----
    gemm_wmma<4, 2><<<dim3(ceil_div(N, 16), 1), 128, 0, stream>>>(
        h2h, nepp, nepb, N, 256, 256, nodein, nlg, nlb, zp, n2g, nullptr, nullptr, nullptr);
    gemm_wmma<2, 1><<<dim3(ceil_div(N, 16), 1), 128, 0, stream>>>(
        nodein, nd1p, nd1b, N, 256, 128, hn, ndg, ndb, nullptr, nullptr, nullptr, nullptr, nullptr);
    k_out_small<<<ceil_div((long long)N * 4, 256), 256, 0, stream>>>(hn, 128, nd2w, nd2b, N, 4, oNode);
}
